// Diffusion_33440615367323
// MI455X (gfx1250) — compile-verified, matrix operand order checked by reading, not run on hardware
//
#include <hip/hip_runtime.h>
#include <hip/hip_bf16.h>

// ---------------------------------------------------------------------------
// EGNN diffusion block for MI455X (gfx1250, wave32, WMMA bf16 16x16x32).
//   B=2, N=192, FEAT=256, FP=257, EIN=516, L=9 layers.
// Edge MLPs restructured: E@W1 = P1[i]+P2[j]+d2*w514+d2i*w515  (node-level),
// remaining per-edge 257x257 GEMMs run on v_wmma_f32_16x16x32_bf16 with
// double-buffered LDS chunks and alternating-register B-tile pipelining.
// ---------------------------------------------------------------------------

#define NL   9
#define NB   2
#define NN   192
#define FP   257           // feature dim (256 + T)
#define KP   288           // K padded to 9*32
#define NP   272           // N padded to 17*16
#define NTIL 17            // N tiles of 16
#define KCH  9             // K chunks of 32
#define NJT  12            // j tiles per row (192/16)
#define LDA  40            // LDS stride (bf16 elems) for A/B tiles

typedef __attribute__((ext_vector_type(16))) __bf16 v16bf;
typedef __attribute__((ext_vector_type(8)))  float  v8f;

__device__ __forceinline__ float silu_f(float x) {
    return x / (1.0f + __expf(-x));
}
__device__ __forceinline__ float sigm_f(float x) {
    return 1.0f / (1.0f + __expf(-x));
}

// ------------------------------ init -----------------------------------
__global__ void k_init(const float* __restrict__ X, const float* __restrict__ H,
                       const float* __restrict__ T, float* __restrict__ Xc,
                       float* __restrict__ Hc) {
    int idx = blockIdx.x * blockDim.x + threadIdx.x;
    if (idx < NB * NN * KP) {
        int bi = idx / KP, f = idx % KP;
        float v = 0.0f;
        if (f < 256) v = H[bi * 256 + f];
        else if (f == 256) v = T[bi];
        Hc[idx] = v;
    }
    if (idx < NB * NN * 3) Xc[idx] = X[idx];
}

__global__ void k_d2init(const float* __restrict__ X, float* __restrict__ d2i) {
    int idx = blockIdx.x * blockDim.x + threadIdx.x;
    if (idx >= NB * NN * NN) return;
    int j = idx % NN, i = (idx / NN) % NN, b = idx / (NN * NN);
    int bi = b * NN + i, bj = b * NN + j;
    float dx = X[bi * 3 + 0] - X[bj * 3 + 0];
    float dy = X[bi * 3 + 1] - X[bj * 3 + 1];
    float dz = X[bi * 3 + 2] - X[bj * 3 + 2];
    d2i[idx] = dx * dx + dy * dy + dz * dz;
}

// Convert msg_w2 / cw_w2 to transposed, padded bf16: w2t[l][n(272)][k(288)]
__global__ void k_wconv(const float* __restrict__ msg_w2,
                        const float* __restrict__ cw_w2,
                        __bf16* __restrict__ w2tm, __bf16* __restrict__ w2tc) {
    int idx = blockIdx.x * blockDim.x + threadIdx.x;
    if (idx >= NL * NP * KP) return;
    int k = idx % KP, n = (idx / KP) % NP, l = idx / (KP * NP);
    bool v = (n < FP) && (k < FP);
    w2tm[idx] = v ? (__bf16)msg_w2[(l * FP + k) * FP + n] : (__bf16)0.0f;
    w2tc[idx] = v ? (__bf16)cw_w2[(l * FP + k) * FP + n] : (__bf16)0.0f;
}

// ---------------------- per-layer node prep (P1,P2,Q1,Q2) ------------------
__global__ __launch_bounds__(256)
void k_prep(int l, const float* __restrict__ Hc,
            const float* __restrict__ msg_w1, const float* __restrict__ msg_b1,
            const float* __restrict__ cw_w1,  const float* __restrict__ cw_b1,
            float* __restrict__ P1, float* __restrict__ P2,
            float* __restrict__ Q1, float* __restrict__ Q2) {
    __shared__ float h[FP];
    int bi = blockIdx.x, tid = threadIdx.x;
    for (int k = tid; k < FP; k += 256) h[k] = Hc[bi * KP + k];
    __syncthreads();
    const float* w1m = msg_w1 + (size_t)l * 516 * FP;
    const float* w1c = cw_w1  + (size_t)l * 516 * FP;
    for (int n = tid; n < KP; n += 256) {
        float p1 = 0.f, p2 = 0.f, q1 = 0.f, q2 = 0.f;
        if (n < FP) {
            p1 = msg_b1[l * FP + n];
            q1 = cw_b1[l * FP + n];
            for (int k = 0; k < FP; ++k) {
                float hv = h[k];
                p1 += hv * w1m[k * FP + n];
                p2 += hv * w1m[(257 + k) * FP + n];
                q1 += hv * w1c[k * FP + n];
                q2 += hv * w1c[(257 + k) * FP + n];
            }
        }
        P1[bi * KP + n] = p1; P2[bi * KP + n] = p2;
        Q1[bi * KP + n] = q1; Q2[bi * KP + n] = q2;
    }
}

// ---------------------------- fused edge kernel ----------------------------
union BV { uint4 q[2]; v16bf v; };

// Generate one K-chunk of the A matrix (silu of rank-reduced preactivation)
// and stage the matching W2^T slice, into LDS buffer `buf`.
__device__ __forceinline__ void gen_chunk(
    int kc, __bf16* __restrict__ Asb, __bf16* __restrict__ Bsb,
    const float* __restrict__ gp1, const float* __restrict__ gp2,
    float gd2, float gd2i, int ge, int gko,
    const float* __restrict__ w514, const float* __restrict__ w515,
    const __bf16* __restrict__ w2t) {
    const int tid = threadIdx.x;
    // A: thread handles half a row: edge ge, k in [kc*32+gko, +16)
    union { uint4 q[2]; __bf16 h[16]; } st;
#pragma unroll
    for (int q = 0; q < 16; ++q) {
        int kg = kc * 32 + gko + q;
        float val = 0.0f;
        if (kg < FP) {
            float x = gp1[kg] + gp2[kg] + gd2 * w514[kg] + gd2i * w515[kg];
            val = silu_f(x);
        }
        st.h[q] = (__bf16)val;
    }
    *(uint4*)(Asb + ge * LDA + gko)     = st.q[0];
    *(uint4*)(Asb + ge * LDA + gko + 8) = st.q[1];
    // B: 272 rows x 32 k, uint4 (8 bf16) copies
    for (int idx = tid; idx < NP * 4; idx += 512) {
        int n = idx >> 2, q4 = idx & 3;
        *(uint4*)(Bsb + n * LDA + q4 * 8) =
            *(const uint4*)(w2t + n * KP + kc * 32 + q4 * 8);
    }
}

// One GEMM+epilogue pass (msg or coord). Wave w owns row i = i0+w (16 edges).
template <bool IS_MSG>
__device__ __forceinline__ void edge_pass(
    int b, int i0, int jt,
    const float* __restrict__ P1, const float* __restrict__ P2,
    const float* __restrict__ w514, const float* __restrict__ w515, // LDS [288]
    const __bf16* __restrict__ w2t,                                 // layer base
    const float* __restrict__ b2v, const float* __restrict__ dotv,  // LDS [272]
    float dot_bias,
    const float* __restrict__ kbuf, const float* __restrict__ d2buf,
    const float* __restrict__ d2ib, const float* __restrict__ dnb,  // LDS
    __bf16* __restrict__ As, __bf16* __restrict__ Bs,   // LDS, 2 buffers each
    float* __restrict__ aggp, float* __restrict__ dXp) {

    const int tid  = threadIdx.x;
    const int wave = tid >> 5, lane = tid & 31;
    const int half = lane >> 4, lm = lane & 15;
    const int j0   = jt * 16;

    // per-thread A-generation assignment (edge ge, half-row gko)
    const int ge  = tid >> 1;
    const int gko = (tid & 1) << 4;
    const float gd2  = d2buf[ge];
    const float gd2i = d2ib[ge];
    const float* gp1 = P1 + (size_t)(b * NN + i0 + (ge >> 4)) * KP;
    const float* gp2 = P2 + (size_t)(b * NN + j0 + (ge & 15)) * KP;

    v8f acc[NTIL];
#pragma unroll
    for (int t = 0; t < NTIL; ++t)
#pragma unroll
        for (int r = 0; r < 8; ++r) acc[t][r] = 0.0f;

    gen_chunk(0, As, Bs, gp1, gp2, gd2, gd2i, ge, gko, w514, w515, w2t);
    __syncthreads();

    const int arow = (wave << 4) + lm;
    for (int kc = 0; kc < KCH; ++kc) {
        const __bf16* Asb = As + (kc & 1) * (256 * LDA);
        const __bf16* Bsb = Bs + (kc & 1) * (NP * LDA);
        union { uint4 q[2]; v16bf v; } af;
        BV ba, bb;   // explicitly alternating B-tile buffers (no copies)
        af.q[0] = *(const uint4*)(Asb + arow * LDA + half * 8);
        af.q[1] = *(const uint4*)(Asb + arow * LDA + 16 + half * 8);
        ba.q[0] = *(const uint4*)(Bsb + (0 * 16 + lm) * LDA + half * 8);
        ba.q[1] = *(const uint4*)(Bsb + (0 * 16 + lm) * LDA + 16 + half * 8);
        bb.q[0] = *(const uint4*)(Bsb + (1 * 16 + lm) * LDA + half * 8);
        bb.q[1] = *(const uint4*)(Bsb + (1 * 16 + lm) * LDA + 16 + half * 8);
#pragma unroll
        for (int t = 0; t < NTIL; ++t) {
            acc[t] = __builtin_amdgcn_wmma_f32_16x16x32_bf16(
                false, af.v, false, (t & 1) ? bb.v : ba.v, (short)0, acc[t],
                false, false);
            if (t + 2 < NTIL) {   // refill the buffer just consumed
                BV& dst = (t & 1) ? bb : ba;
                int brow = (t + 2) * 16 + lm;
                dst.q[0] = *(const uint4*)(Bsb + brow * LDA + half * 8);
                dst.q[1] = *(const uint4*)(Bsb + brow * LDA + 16 + half * 8);
            }
        }
        if (kc + 1 < KCH)
            gen_chunk(kc + 1, As + ((kc + 1) & 1) * (256 * LDA),
                      Bs + ((kc + 1) & 1) * (NP * LDA),
                      gp1, gp2, gd2, gd2i, ge, gko, w514, w515, w2t);
        __syncthreads();
    }

    // ---- epilogue: bias + silu, rank-1 dot over n (butterfly within halves)
    float dotr[8];
#pragma unroll
    for (int r = 0; r < 8; ++r) dotr[r] = 0.0f;
#pragma unroll
    for (int t = 0; t < NTIL; ++t) {
        int n = t * 16 + lm;
        float bb2 = b2v[n], dv = dotv[n];
#pragma unroll
        for (int r = 0; r < 8; ++r) {
            float h = silu_f(acc[t][r] + bb2);  // pads: bias 0 -> silu(0)=0
            acc[t][r] = h;
            dotr[r] += h * dv;                  // dv pads are 0
        }
    }
#pragma unroll
    for (int m = 1; m < 16; m <<= 1)
#pragma unroll
        for (int r = 0; r < 8; ++r) dotr[r] += __shfl_xor(dotr[r], m, 32);

    const int i  = i0 + wave;
    const int bi = b * NN + i;
    float gate[8];
#pragma unroll
    for (int r = 0; r < 8; ++r) {
        int e = (wave << 4) + r + 8 * half;     // edge = 16*i_local + j_local
        float d = dotr[r] + dot_bias;
        gate[r] = IS_MSG ? (kbuf[e] * sigm_f(d)) : (d * kbuf[e]);
    }

    if (IS_MSG) {
        // agg_partial[i, n] = sum_j MH * WH over this j-tile
#pragma unroll
        for (int t = 0; t < NTIL; ++t) {
            float s = 0.0f;
#pragma unroll
            for (int r = 0; r < 8; ++r) s += acc[t][r] * gate[r];
            s += __shfl_xor(s, 16, 32);
            int n = t * 16 + lm;
            if (half == 0 && n < FP)
                aggp[(size_t)(bi * NJT + jt) * NP + n] = s;
        }
    } else {
        float sx = 0.f, sy = 0.f, sz = 0.f;
#pragma unroll
        for (int r = 0; r < 8; ++r) {
            int e = (wave << 4) + r + 8 * half;
            sx += dnb[e * 3 + 0] * gate[r];
            sy += dnb[e * 3 + 1] * gate[r];
            sz += dnb[e * 3 + 2] * gate[r];
        }
        sx += __shfl_xor(sx, 16, 32);
        sy += __shfl_xor(sy, 16, 32);
        sz += __shfl_xor(sz, 16, 32);
        if (lane == 0) {
            float* p = dXp + (size_t)(bi * NJT + jt) * 4;
            p[0] = sx; p[1] = sy; p[2] = sz;
        }
    }
}

__global__ __launch_bounds__(512, 1)
void k_edge(int l, const float* __restrict__ Xc, const float* __restrict__ d2init,
            const float* __restrict__ Kg,
            const float* __restrict__ P1, const float* __restrict__ P2,
            const float* __restrict__ Q1, const float* __restrict__ Q2,
            const __bf16* __restrict__ w2tm, const __bf16* __restrict__ w2tc,
            const float* __restrict__ msg_w1, const float* __restrict__ cw_w1,
            const float* __restrict__ msg_b2, const float* __restrict__ cw_b2,
            const float* __restrict__ fw_w, const float* __restrict__ fw_b,
            const float* __restrict__ cw_w3, const float* __restrict__ cw_b3,
            float* __restrict__ aggp, float* __restrict__ dXp) {
    __shared__ __align__(16) __bf16 As[2 * 256 * LDA];
    __shared__ __align__(16) __bf16 Bs[2 * NP * LDA];
    __shared__ float kbuf[256], d2buf[256], d2ibuf[256];
    __shared__ float dnb[256 * 3];
    __shared__ float w514m[KP], w515m[KP], w514c[KP], w515c[KP];
    __shared__ float b2m[NP], b2c[NP], fwv[NP], c3v[NP];

    const int tid = threadIdx.x;
    const int jt = blockIdx.x, it = blockIdx.y, b = blockIdx.z;
    const int i0 = it * 16, j0 = jt * 16;

    if (tid < 256) {
        int i = i0 + (tid >> 4), j = j0 + (tid & 15);
        int bi = b * NN + i, bj = b * NN + j;
        float dx = Xc[bi * 3 + 0] - Xc[bj * 3 + 0];
        float dy = Xc[bi * 3 + 1] - Xc[bj * 3 + 1];
        float dz = Xc[bi * 3 + 2] - Xc[bj * 3 + 2];
        float d2 = dx * dx + dy * dy + dz * dz;
        d2buf[tid] = d2;
        float inv = 1.0f / (sqrtf(d2) + 1.0f);
        dnb[tid * 3 + 0] = dx * inv;
        dnb[tid * 3 + 1] = dy * inv;
        dnb[tid * 3 + 2] = dz * inv;
        d2ibuf[tid] = d2init[(size_t)bi * NN + j];
        kbuf[tid]   = Kg[(size_t)bi * NN + j];
    }
    const float* w1m = msg_w1 + (size_t)l * 516 * FP;
    const float* w1c = cw_w1  + (size_t)l * 516 * FP;
    for (int n = tid; n < NP; n += 512) {
        bool v = n < FP;
        b2m[n] = v ? msg_b2[l * FP + n] : 0.0f;
        b2c[n] = v ? cw_b2[l * FP + n]  : 0.0f;
        fwv[n] = v ? fw_w[l * FP + n]   : 0.0f;
        c3v[n] = v ? cw_w3[l * FP + n]  : 0.0f;
    }
    for (int k = tid; k < KP; k += 512) {
        bool v = k < FP;
        w514m[k] = v ? w1m[514 * FP + k] : 0.0f;
        w515m[k] = v ? w1m[515 * FP + k] : 0.0f;
        w514c[k] = v ? w1c[514 * FP + k] : 0.0f;
        w515c[k] = v ? w1c[515 * FP + k] : 0.0f;
    }
    __syncthreads();

    // message path -> agg partials
    edge_pass<true>(b, i0, jt, P1, P2, w514m, w515m,
                    w2tm + (size_t)l * NP * KP, b2m, fwv, fw_b[l],
                    kbuf, d2buf, d2ibuf, dnb, As, Bs, aggp, dXp);
    __syncthreads();
    // coord path -> dX partials
    edge_pass<false>(b, i0, jt, Q1, Q2, w514c, w515c,
                     w2tc + (size_t)l * NP * KP, b2c, c3v, cw_b3[l],
                     kbuf, d2buf, d2ibuf, dnb, As, Bs, aggp, dXp);
}

// --------------------------- node update ---------------------------------
__global__ __launch_bounds__(256)
void k_node(int l, float* __restrict__ Hc, float* __restrict__ Xc,
            const float* __restrict__ aggp, const float* __restrict__ dXp,
            const float* __restrict__ up_w1, const float* __restrict__ up_b1,
            const float* __restrict__ up_w2, const float* __restrict__ up_b2) {
    __shared__ float HU[2 * FP];
    __shared__ float u1[FP];
    int bi = blockIdx.x, tid = threadIdx.x;
    for (int n = tid; n < FP; n += 256) {
        HU[n] = Hc[bi * KP + n];
        float s = 0.0f;
        for (int j = 0; j < NJT; ++j) s += aggp[(size_t)(bi * NJT + j) * NP + n];
        HU[FP + n] = s;
    }
    __syncthreads();
    const float* w1 = up_w1 + (size_t)l * 514 * FP;
    for (int n = tid; n < FP; n += 256) {
        float a = up_b1[l * FP + n];
        for (int k = 0; k < 2 * FP; ++k) a += HU[k] * w1[k * FP + n];
        u1[n] = silu_f(a);
    }
    __syncthreads();
    const float* w2 = up_w2 + (size_t)l * FP * FP;
    for (int n = tid; n < FP; n += 256) {
        float a = up_b2[l * FP + n];
        for (int k = 0; k < FP; ++k) a += u1[k] * w2[k * FP + n];
        Hc[bi * KP + n] = a + HU[n];
    }
    if (tid < 3) {
        float s = Xc[bi * 3 + tid];
        for (int j = 0; j < NJT; ++j) s += dXp[(size_t)(bi * NJT + j) * 4 + tid];
        Xc[bi * 3 + tid] = s;
    }
}

// ------------------------------- output ----------------------------------
__global__ void k_final(const float* __restrict__ Xc, const float* __restrict__ Hc,
                        float* __restrict__ out) {
    int idx = blockIdx.x * blockDim.x + threadIdx.x;
    if (idx >= NB * NN * 259) return;
    int f = idx % 259, bi = idx / 259;
    out[idx] = (f < 3) ? Xc[bi * 3 + f] : Hc[bi * KP + (f - 3)];
}

// ------------------------------- launcher --------------------------------
extern "C" void kernel_launch(void* const* d_in, const int* in_sizes, int n_in,
                              void* d_out, int out_size, void* d_ws, size_t ws_size,
                              hipStream_t stream) {
    const float* X      = (const float*)d_in[0];
    const float* H      = (const float*)d_in[1];
    const float* Kg     = (const float*)d_in[2];
    const float* T      = (const float*)d_in[3];
    const float* msg_w1 = (const float*)d_in[4];
    const float* msg_b1 = (const float*)d_in[5];
    const float* msg_w2 = (const float*)d_in[6];
    const float* msg_b2 = (const float*)d_in[7];
    const float* fw_w   = (const float*)d_in[8];
    const float* fw_b   = (const float*)d_in[9];
    const float* up_w1  = (const float*)d_in[10];
    const float* up_b1  = (const float*)d_in[11];
    const float* up_w2  = (const float*)d_in[12];
    const float* up_b2  = (const float*)d_in[13];
    const float* cw_w1  = (const float*)d_in[14];
    const float* cw_b1  = (const float*)d_in[15];
    const float* cw_w2  = (const float*)d_in[16];
    const float* cw_b2  = (const float*)d_in[17];
    const float* cw_w3  = (const float*)d_in[18];
    const float* cw_b3  = (const float*)d_in[19];
    float* out = (float*)d_out;

    char* wp = (char*)d_ws;
    auto alloc = [&](size_t bytes) -> void* {
        void* p = (void*)wp;
        wp += (bytes + 255) & ~(size_t)255;
        return p;
    };
    float*  Xc    = (float*)alloc((size_t)NB * NN * 3 * 4);
    float*  Hc    = (float*)alloc((size_t)NB * NN * KP * 4);
    float*  d2i   = (float*)alloc((size_t)NB * NN * NN * 4);
    float*  P1    = (float*)alloc((size_t)NB * NN * KP * 4);
    float*  P2    = (float*)alloc((size_t)NB * NN * KP * 4);
    float*  Q1    = (float*)alloc((size_t)NB * NN * KP * 4);
    float*  Q2    = (float*)alloc((size_t)NB * NN * KP * 4);
    float*  aggp  = (float*)alloc((size_t)NB * NN * NJT * NP * 4);
    float*  dXp   = (float*)alloc((size_t)NB * NN * NJT * 4 * 4);
    __bf16* w2tm  = (__bf16*)alloc((size_t)NL * NP * KP * 2);
    __bf16* w2tc  = (__bf16*)alloc((size_t)NL * NP * KP * 2);
    (void)ws_size; (void)in_sizes; (void)n_in; (void)out_size;

    k_init<<<(NB * NN * KP + 255) / 256, 256, 0, stream>>>(X, H, T, Xc, Hc);
    k_d2init<<<(NB * NN * NN + 255) / 256, 256, 0, stream>>>(X, d2i);
    k_wconv<<<(NL * NP * KP + 255) / 256, 256, 0, stream>>>(msg_w2, cw_w2, w2tm, w2tc);

    for (int l = 0; l < NL; ++l) {
        k_prep<<<NB * NN, 256, 0, stream>>>(l, Hc, msg_w1, msg_b1, cw_w1, cw_b1,
                                            P1, P2, Q1, Q2);
        k_edge<<<dim3(NJT, NN / 16, NB), 512, 0, stream>>>(
            l, Xc, d2i, Kg, P1, P2, Q1, Q2, w2tm, w2tc,
            msg_w1, cw_w1, msg_b2, cw_b2, fw_w, fw_b, cw_w3, cw_b3, aggp, dXp);
        k_node<<<NB * NN, 256, 0, stream>>>(l, Hc, Xc, aggp, dXp,
                                            up_w1, up_b1, up_w2, up_b2);
    }
    k_final<<<(NB * NN * 259 + 255) / 256, 256, 0, stream>>>(Xc, Hc, out);
}
